// MultiHeadSelfAttention_87436944212312
// MI455X (gfx1250) — matrix operand order repaired and checked by reading, not compiled, not run on hardware
//
#include <hip/hip_runtime.h>
#include <hip/hip_bf16.h>
#include <math.h>

#define DEV __device__ __forceinline__

typedef __attribute__((ext_vector_type(16))) __bf16 v16bf;
typedef __attribute__((ext_vector_type(8)))  float  v8f;
typedef __attribute__((ext_vector_type(16))) unsigned short v16us;
typedef __attribute__((ext_vector_type(8)))  unsigned short v8us;
typedef __attribute__((ext_vector_type(4)))  unsigned short v4us;
typedef __attribute__((ext_vector_type(4)))  float  f4;

union BFrag { v16us u; v16bf b; unsigned short s[16]; };

static constexpr int L_SEQ = 2048;
static constexpr int NBATCH = 4;
static constexpr int E_DIM = 1024;
static constexpr int H_NUM = 16;
static constexpr int DH = 64;
static constexpr int BH = NBATCH * H_NUM;   // 64 batch-heads
static constexpr int ROWS = L_SEQ * NBATCH; // 8192
static constexpr int N_QKV = 3 * E_DIM;     // 3072

DEV unsigned short f2bf(float f) {
  unsigned int x = __float_as_uint(f);
  x += 0x7FFFu + ((x >> 16) & 1u);   // round-to-nearest-even
  return (unsigned short)(x >> 16);
}

DEV v8f wmma_bf16(v16bf a, v16bf b, v8f c) {
  return __builtin_amdgcn_wmma_f32_16x16x32_bf16(false, a, false, b, (short)0, c, false, false);
}

// A-operand fragment (16 rows x 32 K): lanes 0-15 row=lane, K = kb+0..15;
// lanes 16-31 same row (lane&15), K = kb+16..31.
DEV v16bf load_a16(const unsigned short* p0, int stride, int kbase, int lane) {
  int r  = lane & 15;
  int kh = kbase + ((lane & 16) ? 16 : 0);
  const unsigned short* p = p0 + r * stride + kh;
  BFrag t;
  *(v8us*)&t.s[0] = *(const v8us*)(p);
  *(v8us*)&t.s[8] = *(const v8us*)(p + 8);
  return t.b;
}

// B-operand fragment (32 K x 16 cols): lanes 0-15 col=lane, K = kb+0..15;
// lanes 16-31 col=lane-16, K = kb+16..31.
DEV v16bf load_b16(const unsigned short* p0, int stride, int kbase, int lane) {
  int c  = lane & 15;
  int kh = kbase + ((lane & 16) ? 16 : 0);
  const unsigned short* p = p0 + c * stride + kh;
  BFrag t;
  *(v8us*)&t.s[0] = *(const v8us*)(p);
  *(v8us*)&t.s[8] = *(const v8us*)(p + 8);
  return t.b;
}

// Per-lane prefetch of the row this lane will load on a later K-step.
DEV void prefetch_row16(const unsigned short* p0, int stride, int kbase, int lane) {
  int r = lane & 15;
  __builtin_prefetch((const void*)(p0 + r * stride + kbase), 0, 3);
}

// Vectorized fp32 -> bf16 conversion (4 elements/thread/iter).
__global__ void cvt_f32_bf16_x4(const float* __restrict__ in,
                                unsigned short* __restrict__ out, int n4) {
  int i = blockIdx.x * blockDim.x + threadIdx.x;
  int stride = gridDim.x * blockDim.x;
  for (; i < n4; i += stride) {
    f4 v = *(const f4*)(in + 4 * i);
    v4us o;
    o.x = f2bf(v.x); o.y = f2bf(v.y); o.z = f2bf(v.z); o.w = f2bf(v.w);
    *(v4us*)(out + 4 * i) = o;
  }
}

// Software-pipelined 32x64-per-wave GEMM core: C[2][4] += A(rows) * B(cols)^T.
// Fragments for step k+32 are loaded before the step-k WMMAs are issued, so
// the WMMAs never wait on the most recent loads. Unrolled x2 so the two
// fragment buffers ping-pong by register renaming instead of v_mov copies.
DEV void gemm_core(const unsigned short* __restrict__ A0,
                   const unsigned short* __restrict__ A1,
                   const unsigned short* __restrict__ Wc, // W + col0*E
                   v8f acc[2][4], int lane) {
  v16bf a0 = load_a16(A0, E_DIM, 0, lane);
  v16bf a1 = load_a16(A1, E_DIM, 0, lane);
  v16bf b[4], bn[4];
#pragma unroll
  for (int j = 0; j < 4; ++j) b[j] = load_b16(Wc + j * 16 * E_DIM, E_DIM, 0, lane);

#pragma unroll 2
  for (int k = 0; k < E_DIM; k += 32) {
    int kn = (k + 32) & (E_DIM - 1);   // wrap on last step; result discarded
    v16bf a0n = load_a16(A0, E_DIM, kn, lane);
    v16bf a1n = load_a16(A1, E_DIM, kn, lane);
#pragma unroll
    for (int j = 0; j < 4; ++j) bn[j] = load_b16(Wc + j * 16 * E_DIM, E_DIM, kn, lane);
    prefetch_row16(A0, E_DIM, (k + 64) & (E_DIM - 1), lane);
    prefetch_row16(A1, E_DIM, (k + 64) & (E_DIM - 1), lane);

#pragma unroll
    for (int j = 0; j < 4; ++j) {
      acc[0][j] = wmma_bf16(a0, b[j], acc[0][j]);
      acc[1][j] = wmma_bf16(a1, b[j], acc[1][j]);
    }
    a0 = a0n; a1 = a1n;
#pragma unroll
    for (int j = 0; j < 4; ++j) b[j] = bn[j];
  }
}

// qkv = Xbf @ Wqkv^T  (both bf16); epilogue scatters into head-major bf16 Q
// (scaled by Dh^-0.5), K, and transposed Vt [Dh, L].
__global__ __launch_bounds__(256)
void qkv_gemm(const unsigned short* __restrict__ X, const unsigned short* __restrict__ W,
              unsigned short* __restrict__ Q, unsigned short* __restrict__ Kh,
              unsigned short* __restrict__ Vt) {
  int lane = threadIdx.x & 31;
  int wave = threadIdx.x >> 5;
  int row0 = blockIdx.x * 128 + (wave & 3) * 32;
  int col0 = blockIdx.y * 128 + (wave >> 2) * 64;

  v8f acc[2][4] = {};
  gemm_core(X + row0 * E_DIM, X + (row0 + 16) * E_DIM, W + col0 * E_DIM, acc, lane);

  int hi8 = (lane & 16) ? 8 : 0;
  int cl  = lane & 15;
#pragma unroll
  for (int i = 0; i < 2; ++i)
#pragma unroll
    for (int j = 0; j < 4; ++j) {
      const float* av = (const float*)&acc[i][j];
#pragma unroll
      for (int r = 0; r < 8; ++r) {
        int m   = row0 + i * 16 + r + hi8;
        int col = col0 + j * 16 + cl;
        float v = av[r];
        int l  = m >> 2, nb = m & 3;
        int part = col >> 10, e = col & 1023;
        int h = e >> 6, d = e & 63;
        int bh = nb * H_NUM + h;
        if (part == 0)      Q[(bh * L_SEQ + l) * DH + d] = f2bf(v * 0.125f);
        else if (part == 1) Kh[(bh * L_SEQ + l) * DH + d] = f2bf(v);
        else                Vt[(bh * DH + d) * L_SEQ + l] = f2bf(v);
      }
    }
}

// Flash attention: one wave = 16 queries. Computes S^T = K*Q^T (softmax reduction
// stays in-lane), online softmax, O^T = V^T * P^T. Causal mask applied analytically.
// K/V fragments for the next key block are preloaded at the top of each iteration.
__global__ __launch_bounds__(256)
void attn_fa(const unsigned short* __restrict__ Q, const unsigned short* __restrict__ Kh,
             const unsigned short* __restrict__ Vt, unsigned short* __restrict__ O) {
  int lane = threadIdx.x & 31;
  int wave = threadIdx.x >> 5;
  int bh = blockIdx.y;
  int nb = bh >> 4, h = bh & 15;
  int q0 = blockIdx.x * 128 + wave * 16;

  const unsigned short* Qb = Q  + (size_t)bh * L_SEQ * DH;
  const unsigned short* Kb = Kh + (size_t)bh * L_SEQ * DH;
  const unsigned short* Vb = Vt + (size_t)bh * DH * L_SEQ;

  v16bf qb0 = load_b16(Qb + q0 * DH, DH, 0, lane);
  v16bf qb1 = load_b16(Qb + q0 * DH, DH, 32, lane);

  v8f o[4] = {};
  float m_run = -INFINITY, l_run = 0.f;
  int qg  = q0 + (lane & 15);
  int hi8 = (lane & 16) ? 8 : 0;

  int kend = ((q0 + 16 + 31) >> 5) << 5;   // last key block touching this q-tile

  // preload key block 0
  v16bf kf[4], vf[4];
  kf[0] = load_a16(Kb, DH, 0, lane);
  kf[1] = load_a16(Kb, DH, 32, lane);
  kf[2] = load_a16(Kb + 16 * DH, DH, 0, lane);
  kf[3] = load_a16(Kb + 16 * DH, DH, 32, lane);
#pragma unroll
  for (int t = 0; t < 4; ++t) vf[t] = load_a16(Vb + (t * 16) * L_SEQ, L_SEQ, 0, lane);

#pragma unroll 2
  for (int kb = 0; kb < kend; kb += 32) {
    int kbn = (kb + 32 < kend) ? kb + 32 : kb;   // clamp: last preload discarded
    v16bf kfn[4], vfn[4];
    kfn[0] = load_a16(Kb + kbn * DH, DH, 0, lane);
    kfn[1] = load_a16(Kb + kbn * DH, DH, 32, lane);
    kfn[2] = load_a16(Kb + (kbn + 16) * DH, DH, 0, lane);
    kfn[3] = load_a16(Kb + (kbn + 16) * DH, DH, 32, lane);
#pragma unroll
    for (int t = 0; t < 4; ++t) vfn[t] = load_a16(Vb + (t * 16) * L_SEQ + kbn, L_SEQ, 0, lane);
    prefetch_row16(Kb + (kbn + 32) * DH, DH, 0, lane);

    v8f st0 = {}, st1 = {};
    st0 = wmma_bf16(kf[0], qb0, st0);
    st0 = wmma_bf16(kf[1], qb1, st0);
    st1 = wmma_bf16(kf[2], qb0, st1);
    st1 = wmma_bf16(kf[3], qb1, st1);

    float* s0 = (float*)&st0;
    float* s1 = (float*)&st1;
    float mloc = -INFINITY;
#pragma unroll
    for (int r = 0; r < 8; ++r) {
      int k0 = kb + hi8 + r;
      if (k0 > qg)      s0[r] = -100000.0f;
      if (k0 + 16 > qg) s1[r] = -100000.0f;
      mloc = fmaxf(mloc, fmaxf(s0[r], s1[r]));
    }
    mloc = fmaxf(mloc, __shfl_xor(mloc, 16, 32));
    float mnew  = fmaxf(m_run, mloc);
    float alpha = __expf(m_run - mnew);

    // lane pair exchange so each lane holds a contiguous 16-key column of P^T
    float x0[8], x1[8];
#pragma unroll
    for (int r = 0; r < 8; ++r) {
      x0[r] = __shfl_xor(s0[r], 16, 32);
      x1[r] = __shfl_xor(s1[r], 16, 32);
    }
    BFrag pb;
    float lloc = 0.f;
#pragma unroll
    for (int r = 0; r < 8; ++r) {
      float eA = __expf(((lane & 16) ? x1[r] : s0[r]) - mnew);
      float eB = __expf(((lane & 16) ? s1[r] : x0[r]) - mnew);
      lloc += eA + eB;
      pb.s[r]     = f2bf(eA);
      pb.s[r + 8] = f2bf(eB);
    }
    lloc  += __shfl_xor(lloc, 16, 32);
    l_run  = l_run * alpha + lloc;
    m_run  = mnew;

#pragma unroll
    for (int t = 0; t < 4; ++t) {
      o[t] = o[t] * alpha;
      o[t] = wmma_bf16(vf[t], pb.b, o[t]);
    }
#pragma unroll
    for (int t = 0; t < 4; ++t) { kf[t] = kfn[t]; vf[t] = vfn[t]; }
  }

  float inv = 1.0f / l_run;
#pragma unroll
  for (int t = 0; t < 4; ++t) {
    const float* ov = (const float*)&o[t];
#pragma unroll
    for (int r = 0; r < 8; ++r) {
      int dg  = t * 16 + r + hi8;
      int row = qg * NBATCH + nb;          // [L, N, E] layout
      int col = h * DH + dg;
      O[row * E_DIM + col] = f2bf(ov[r] * inv);
    }
  }
}

// out = O @ Wout^T  (O bf16 [8192,1024], W bf16 [1024,1024]) -> fp32
__global__ __launch_bounds__(256)
void out_gemm(const unsigned short* __restrict__ A, const unsigned short* __restrict__ W,
              float* __restrict__ C) {
  int lane = threadIdx.x & 31;
  int wave = threadIdx.x >> 5;
  int row0 = blockIdx.x * 128 + (wave & 3) * 32;
  int col0 = blockIdx.y * 128 + (wave >> 2) * 64;

  v8f acc[2][4] = {};
  gemm_core(A + row0 * E_DIM, A + (row0 + 16) * E_DIM, W + col0 * E_DIM, acc, lane);

  int hi8 = (lane & 16) ? 8 : 0;
  int cl  = lane & 15;
#pragma unroll
  for (int i = 0; i < 2; ++i)
#pragma unroll
    for (int j = 0; j < 4; ++j) {
      const float* av = (const float*)&acc[i][j];
#pragma unroll
      for (int r = 0; r < 8; ++r) {
        int m   = row0 + i * 16 + r + hi8;
        int col = col0 + j * 16 + cl;
        C[m * E_DIM + col] = av[r];
      }
    }
}

extern "C" void kernel_launch(void* const* d_in, const int* in_sizes, int n_in,
                              void* d_out, int out_size, void* d_ws, size_t ws_size,
                              hipStream_t stream) {
  const float* x     = (const float*)d_in[0];
  const float* w_qkv = (const float*)d_in[1];
  const float* w_out = (const float*)d_in[2];
  // d_in[3] = boolean causal mask — applied analytically inside attn_fa.

  char* ws = (char*)d_ws;
  unsigned short* Xbf  = (unsigned short*)(ws);             // 16 MiB [8192,1024]
  unsigned short* Wqkv = (unsigned short*)(ws + 16777216);  //  6 MiB [3072,1024]
  unsigned short* Wout = (unsigned short*)(ws + 23068672);  //  2 MiB [1024,1024]
  unsigned short* Qbf  = (unsigned short*)(ws + 25165824);  // 16 MiB [64,2048,64]
  unsigned short* Kbf  = (unsigned short*)(ws + 41943040);  // 16 MiB [64,2048,64]
  unsigned short* Vt   = (unsigned short*)(ws + 58720256);  // 16 MiB [64,64,2048]
  unsigned short* Obf  = (unsigned short*)(ws + 75497472);  // 16 MiB [8192,1024]
  float* out = (float*)d_out;

  cvt_f32_bf16_x4<<<2048, 256, 0, stream>>>(x, Xbf, ROWS * E_DIM / 4);
  cvt_f32_bf16_x4<<<1024, 256, 0, stream>>>(w_qkv, Wqkv, N_QKV * E_DIM / 4);
  cvt_f32_bf16_x4<<<512, 256, 0, stream>>>(w_out, Wout, E_DIM * E_DIM / 4);
  qkv_gemm<<<dim3(ROWS / 128, N_QKV / 128), 256, 0, stream>>>(Xbf, Wqkv, Qbf, Kbf, Vt);
  attn_fa<<<dim3(L_SEQ / 128, BH), 256, 0, stream>>>(Qbf, Kbf, Vt, Obf);
  out_gemm<<<dim3(ROWS / 128, E_DIM / 128), 256, 0, stream>>>(Obf, Wout, out);
}
